// SelfAttentionPatch_83760452206642
// MI455X (gfx1250) — matrix operand order repaired and checked by reading, not compiled, not run on hardware
//
#include <hip/hip_runtime.h>
#include <stdint.h>

// ---------- types ----------
typedef __attribute__((ext_vector_type(16))) _Float16 v16h;
typedef __attribute__((ext_vector_type(8)))  float    v8f;
typedef __attribute__((ext_vector_type(4)))  float    v4f;
typedef __attribute__((ext_vector_type(4)))  uint32_t v4u;

union Frag16 {
  v16h     v;
  _Float16 h[16];
  uint16_t u[16];
  uint32_t w[8];
  v4u      q[2];
};

__device__ __forceinline__ uint16_t f2h(float a) {
  union { _Float16 h; uint16_t u; } t;
  t.h = (_Float16)a;
  return t.u;
}
__device__ __forceinline__ uint32_t pack2(float a, float b) {
  union { _Float16 h[2]; uint32_t w; } t;
  t.h[0] = (_Float16)a; t.h[1] = (_Float16)b;
  return t.w;
}

#define C_IN   128
#define CQK    16
#define HFULL  256
#define WFULL  256
#define NPOOL  4096
#define WROWS  160
#define WPITCH 136   // weight LDS pitch: 68-word rows -> conflict-free b128 A reads
#define XPITCH 264   // x-swizzle LDS pitch: 132-word rows -> conflict-free b128 B reads

// =====================================================================
// Kernel 1: fused conv1x1(q|k|v) + 4x4 maxpool.
// x tile is staged in LDS PRE-SWIZZLED into per-lane WMMA B-fragment
// order: xsw[(kk*32+lane)*XPITCH + j*16 + i]  ->  B frag = 2x ds_load_b128.
// One WMMA N-tile (16 pixels) == one pooling window; max over N via shfl.
// grid: 4 b * 64 ph * 4 col-groups = 1024 blocks of 256 (8 waves).
// =====================================================================
__global__ __launch_bounds__(256) void k_conv_pool(
    const float* __restrict__ x,
    const float* __restrict__ wq, const float* __restrict__ bq,
    const float* __restrict__ wk, const float* __restrict__ bk,
    const float* __restrict__ wv, const float* __restrict__ bv,
    uint16_t* __restrict__ qb, uint16_t* __restrict__ kswz,
    uint16_t* __restrict__ vb)
{
  extern __shared__ char smem[];
  uint16_t* xsw  = (uint16_t*)smem;                 // [128][XPITCH] swizzled x
  uint16_t* wgt  = xsw + 128 * XPITCH;              // [160][WPITCH] weights
  float*    bias = (float*)(wgt + WROWS * WPITCH);  // [160]

  const int tid = threadIdx.x;
  const int gid = blockIdx.x;
  const int b   = gid >> 8;
  const int rem = gid & 255;
  const int ph  = rem >> 2;   // pooled row 0..63
  const int pg  = rem & 3;    // group of 16 pooled cols

  // ---- stage x tile: coalesced float4 reads, swizzled f16 stores ----
  {
    const float* xb = x + (size_t)b * C_IN * (HFULL * WFULL);
    for (int it = 0; it < 32; ++it) {
      int ch = tid + it * 256;          // 8192 float4 chunks
      int c  = ch >> 6;
      int r2 = ch & 63;
      int dy = r2 >> 4;
      int j  = r2 & 15;                 // window (pooled col within group)
      const float* src = xb + ((size_t)c * HFULL + (ph * 4 + dy)) * WFULL
                            + pg * 64 + j * 4;
      v4f f = *(const v4f*)src;
      uint16_t hv[4] = { f2h(f.x), f2h(f.y), f2h(f.z), f2h(f.w) };
      const int kk    = c >> 5;
      const int lhi_d = (c >> 4) & 1;
      const int i     = c & 15;
#pragma unroll
      for (int u = 0; u < 4; ++u) {
        int lane_d = lhi_d * 16 + dy * 4 + u;   // B-operand lane that owns it
        xsw[(kk * 32 + lane_d) * XPITCH + j * 16 + i] = hv[u];
      }
    }
  }
  // ---- stage weights: rows 0-15 wq, 16-31 wk, 32-159 wv ----
  for (int it = 0; it < 80; ++it) {
    int ch  = tid + it * 256;           // 20480 elements
    int r   = ch >> 7;
    int col = ch & 127;
    float wv_;
    if (r < 16)      wv_ = wq[r * 128 + col];
    else if (r < 32) wv_ = wk[(r - 16) * 128 + col];
    else             wv_ = wv[(r - 32) * 128 + col];
    wgt[r * WPITCH + col] = f2h(wv_);
  }
  if (tid < WROWS)
    bias[tid] = (tid < 16) ? bq[tid] : (tid < 32 ? bk[tid - 16] : bv[tid - 32]);
  __syncthreads();

  const int wave = tid >> 5;
  const int lane = tid & 31;
  const int lrow = lane & 15;
  const int lhi  = lane >> 4;

  for (int jj = 0; jj < 2; ++jj) {
    const int j = wave * 2 + jj;        // this wave's window (pooled pixel)
    v8f acc[10] = {};
    for (int kk = 0; kk < 4; ++kk) {
      Frag16 bf;                        // pre-swizzled: contiguous per lane
      const uint16_t* br = &xsw[(kk * 32 + lane) * XPITCH + j * 16];
      bf.q[0] = *(const v4u*)br;
      bf.q[1] = *(const v4u*)(br + 8);
      const int kbA = kk * 32 + lhi * 16;
#pragma unroll
      for (int mt = 0; mt < 10; ++mt) {
        Frag16 af;
        const uint16_t* wr = &wgt[(mt * 16 + lrow) * WPITCH + kbA];
        af.q[0] = *(const v4u*)wr;
        af.q[1] = *(const v4u*)(wr + 8);
        acc[mt] = __builtin_amdgcn_wmma_f32_16x16x32_f16(
            false, af.v, false, bf.v, (short)0, acc[mt], false, false);
      }
    }
    // ---- maxpool over the 16 pixels (N dim -> lanes), add bias, store f16 ----
    const int m = ph * 64 + pg * 16 + j;   // pooled flat index
#pragma unroll
    for (int mt = 0; mt < 10; ++mt) {
#pragma unroll
      for (int r = 0; r < 8; ++r) {
        float v = acc[mt][r];
        v = fmaxf(v, __shfl_xor(v, 1));
        v = fmaxf(v, __shfl_xor(v, 2));
        v = fmaxf(v, __shfl_xor(v, 4));
        v = fmaxf(v, __shfl_xor(v, 8));
        if (lrow == 0) {
          int c = mt * 16 + r + 8 * lhi;
          uint16_t hv = f2h(v + bias[c]);
          if (c < CQK) {
            qb[((size_t)b * NPOOL + m) * CQK + c] = hv;
          } else if (c < 2 * CQK) {
            // k stored directly in per-lane B-fragment order for kernel 2:
            // [n-tile][lane 0..15][i 0..15]
            kswz[(size_t)b * (CQK * NPOOL)
                 + ((m >> 4) * 256 + (m & 15) * 16 + (c - CQK))] = hv;
          } else {
            vb[((size_t)b * C_IN + (c - 2 * CQK)) * NPOOL + m] = hv;
          }
        }
      }
    }
  }
}

// =====================================================================
// Kernel 2: energy = q k^T + exact softmax, two-pass online (recompute
// WMMA instead of buffering 256KB of S). grid: 4b * 32 = 128 blocks,
// 8 waves each own one 16-row m-tile. k[b] (128KB, already in fragment
// order) staged once per block into LDS via async global->LDS copies.
// =====================================================================
__device__ __forceinline__ void load_k_bfrag(const uint16_t* ks, int n0,
                                             int lane, Frag16& bf) {
  {
    const uint16_t* p = ks + ((n0 >> 4) * 256 + (lane & 15) * 16 + (lane >> 4) * 8);
    bf.q[0] = *(const v4u*)p;
#pragma unroll
    for (int i = 4; i < 8; ++i) bf.w[i] = 0u;   // K rows 16..31 are padding
  }
}

__global__ __launch_bounds__(256) void k_attn_softmax(
    const uint16_t* __restrict__ qb, const uint16_t* __restrict__ kswz,
    float* __restrict__ attn)
{
  extern __shared__ char smem2[];
  uint16_t* ks = (uint16_t*)smem2;            // 128KB: k[b] in fragment order
  const int tid  = threadIdx.x;
  const int b    = blockIdx.x >> 5;
  const int mt   = (blockIdx.x & 31) * 8 + (tid >> 5);
  const int lane = tid & 31;
  const int lrow = lane & 15;
  const int lhi  = lane >> 4;

  { // ---- async copy k[b] (8192 x 16B) global -> LDS, ASYNCcnt tracked ----
    const char* src = (const char*)(kswz + (size_t)b * CQK * NPOOL);
    uint32_t lds_base = (uint32_t)(uintptr_t)ks;
    for (int it = 0; it < 32; ++it) {
      uint32_t off = (uint32_t)(tid + it * 256) * 16u;
      uint32_t dst = lds_base + off;
      asm volatile("global_load_async_to_lds_b128 %0, %1, %2 offset:0"
                   :: "v"(dst), "v"(off), "s"(src) : "memory");
    }
    asm volatile("s_wait_asynccnt 0" ::: "memory");
  }
  __syncthreads();

  // A fragment: q rows, K 0..15 real, 16..31 zero-padded
  Frag16 af;
  {
    const int m = mt * 16 + lrow;
    af.q[0] = *(const v4u*)(qb + ((size_t)b * NPOOL + m) * CQK + lhi * 8);
    af.w[4] = 0u; af.w[5] = 0u; af.w[6] = 0u; af.w[7] = 0u;
  }

  float mx[8], sm[8];
#pragma unroll
  for (int r = 0; r < 8; ++r) { mx[r] = -3.0e38f; sm[r] = 0.0f; }

  // ---- pass 1: online row max / exp-sum ----
  for (int n0 = 0; n0 < NPOOL; n0 += 16) {
    Frag16 bf; load_k_bfrag(ks, n0, lane, bf);
    v8f cz = {};
    v8f d = __builtin_amdgcn_wmma_f32_16x16x32_f16(
        false, af.v, false, bf.v, (short)0, cz, false, false);
#pragma unroll
    for (int r = 0; r < 8; ++r) {
      float v  = d[r];
      float m2 = fmaxf(mx[r], v);
      sm[r] = sm[r] * __expf(mx[r] - m2) + __expf(v - m2);
      mx[r] = m2;
    }
  }
  // combine the 16 lanes of each half-wave (same M row group)
#pragma unroll
  for (int r = 0; r < 8; ++r) {
#pragma unroll
    for (int s = 1; s < 16; s <<= 1) {
      float omx = __shfl_xor(mx[r], s);
      float osm = __shfl_xor(sm[r], s);
      float M   = fmaxf(mx[r], omx);
      sm[r] = sm[r] * __expf(mx[r] - M) + osm * __expf(omx - M);
      mx[r] = M;
    }
    sm[r] = 1.0f / sm[r];
  }

  // ---- pass 2: recompute energies, write normalized softmax (fp32) ----
  for (int n0 = 0; n0 < NPOOL; n0 += 16) {
    Frag16 bf; load_k_bfrag(ks, n0, lane, bf);
    v8f cz = {};
    v8f d = __builtin_amdgcn_wmma_f32_16x16x32_f16(
        false, af.v, false, bf.v, (short)0, cz, false, false);
#pragma unroll
    for (int r = 0; r < 8; ++r) {
      int mrow = mt * 16 + r + 8 * lhi;
      attn[((size_t)b * NPOOL + mrow) * NPOOL + n0 + lrow] =
          __expf(d[r] - mx[r]) * sm[r];
    }
  }
}

// =====================================================================
// Kernel 3: out[c,m] = sum_n v[c,n]*attn[m,n], fused 4x4 nearest upsample
// + gamma*out + x. grid: 4b * 64 n-blocks(64 m) = 256 blocks; each of 8
// waves owns a 16-channel M-tile. attn tile staged fp32->f16 in LDS.
// =====================================================================
__global__ __launch_bounds__(256) void k_out_gemm(
    const uint16_t* __restrict__ vb, const float* __restrict__ attn,
    const float* __restrict__ x, const float* __restrict__ gamma,
    float* __restrict__ out)
{
  __shared__ uint16_t att_s[64 * 40];   // 64 m rows, 32 n + pad -> no bank conflicts
  const int tid    = threadIdx.x;
  const int b      = blockIdx.x >> 6;
  const int m_base = (blockIdx.x & 63) * 64;
  const int wave   = tid >> 5;
  const int lane   = tid & 31;
  const int lrow   = lane & 15;
  const int lhi    = lane >> 4;
  const int c0     = wave * 16;

  v8f acc[4] = {};
  const float*    attb = attn + (size_t)b * NPOOL * NPOOL;
  const uint16_t* vbb  = vb + (size_t)b * C_IN * NPOOL;

  for (int k0 = 0; k0 < NPOOL; k0 += 32) {
    __syncthreads();
#pragma unroll
    for (int rep = 0; rep < 2; ++rep) {
      int ch = tid + rep * 256;          // 512 float4 chunks = 64m x 32n
      int ml = ch >> 3;
      int no = (ch & 7) * 4;
      const float* src = attb + (size_t)(m_base + ml) * NPOOL + k0 + no;
      v4f f = *(const v4f*)src;
      uint32_t* d = (uint32_t*)&att_s[ml * 40 + no];
      d[0] = pack2(f.x, f.y);
      d[1] = pack2(f.z, f.w);
    }
    if (k0 + 32 < NPOOL) {               // prefetch next attention tile
      int ml = tid >> 3;
      int no = (tid & 7) * 4;
      __builtin_prefetch(attb + (size_t)(m_base + ml) * NPOOL + (k0 + 32) + no, 0, 1);
    }
    __syncthreads();

    // A fragment: v[c][n] slice for this wave's channel tile
    Frag16 af;
    const uint16_t* vr = vbb + (size_t)(c0 + lrow) * NPOOL + k0 + lhi * 8;
    af.q[0] = *(const v4u*)vr;
    af.q[1] = *(const v4u*)(vr + 16);
#pragma unroll
    for (int t = 0; t < 4; ++t) {
      Frag16 bf;                          // B = attn^T tile from LDS
      const uint16_t* ar = &att_s[(t * 16 + lrow) * 40 + lhi * 8];
      bf.q[0] = *(const v4u*)ar;
      bf.q[1] = *(const v4u*)(ar + 16);
      acc[t] = __builtin_amdgcn_wmma_f32_16x16x32_f16(
          false, af.v, false, bf.v, (short)0, acc[t], false, false);
    }
  }

  // ---- epilogue: 4x4 upsample + gamma*val + x, coalesced float4 stores ----
  const float g = gamma[0];
#pragma unroll
  for (int t = 0; t < 4; ++t) {
#pragma unroll
    for (int r = 0; r < 8; ++r) {
      float val = g * acc[t][r];
      int c  = c0 + r + 8 * lhi;
      int m  = m_base + t * 16 + lrow;
      int pr = m >> 6, pc = m & 63;
      size_t base = ((size_t)(b * C_IN + c) * HFULL + pr * 4) * WFULL + pc * 4;
#pragma unroll
      for (int dy = 0; dy < 4; ++dy) {
        size_t idx = base + (size_t)dy * WFULL;
        v4f xv = *(const v4f*)(x + idx);
        *(v4f*)(out + idx) = xv + val;
      }
    }
  }
}

// =====================================================================
extern "C" void kernel_launch(void* const* d_in, const int* in_sizes, int n_in,
                              void* d_out, int out_size, void* d_ws, size_t ws_size,
                              hipStream_t stream) {
  (void)in_sizes; (void)n_in; (void)out_size; (void)ws_size;
  const float* x     = (const float*)d_in[0];
  const float* wq    = (const float*)d_in[1];
  const float* bq    = (const float*)d_in[2];
  const float* wk    = (const float*)d_in[3];
  const float* bk    = (const float*)d_in[4];
  const float* wv    = (const float*)d_in[5];
  const float* bv    = (const float*)d_in[6];
  const float* gamma = (const float*)d_in[7];

  float* out  = (float*)d_out;
  float* attn = out + (size_t)4 * 128 * 256 * 256;   // second tuple output

  uint16_t* qb = (uint16_t*)d_ws;                    // [4][4096][16] f16
  uint16_t* kb = qb + (size_t)4 * NPOOL * CQK;       // [4][256][16][16] f16 (frag order)
  uint16_t* vb = kb + (size_t)4 * CQK * NPOOL;       // [4][128][4096] f16

  const size_t smem1 = (size_t)(128 * XPITCH + WROWS * WPITCH) * 2 + WROWS * 4;
  const size_t smem2 = (size_t)CQK * NPOOL * 2;
  hipFuncSetAttribute(reinterpret_cast<const void*>(k_conv_pool),
                      hipFuncAttributeMaxDynamicSharedMemorySize, (int)smem1);
  hipFuncSetAttribute(reinterpret_cast<const void*>(k_attn_softmax),
                      hipFuncAttributeMaxDynamicSharedMemorySize, (int)smem2);

  k_conv_pool<<<dim3(1024), dim3(256), smem1, stream>>>(
      x, wq, bq, wk, bk, wv, bv, qb, kb, vb);
  k_attn_softmax<<<dim3(128), dim3(256), smem2, stream>>>(qb, kb, attn);
  k_out_gemm<<<dim3(256), dim3(256), 0, stream>>>(vb, attn, x, gamma, out);
}
